// LinearHijack_41592463295012
// MI455X (gfx1250) — matrix operand order repaired and checked by reading, not compiled, not run on hardware
//
#include <hip/hip_runtime.h>

typedef __attribute__((ext_vector_type(16))) _Float16 v16h;
typedef __attribute__((ext_vector_type(8)))  _Float16 v8h;
typedef __attribute__((ext_vector_type(8)))  float    v8f;
typedef __attribute__((ext_vector_type(4)))  float    f4v;
typedef __attribute__((ext_vector_type(4)))  int      i4v;

#define M_TOT 8192   // BATCH * SEQ
#define N_TOT 4096   // OUT
#define K_TOT 4096   // IN
#define QBLK  64     // absmax block size (== BK)

#define BM 128
#define BN 128
#define BK 64
#define LDSS 72      // padded LDS row stride in halves (144 B, 16B-aligned rows)

__global__ __launch_bounds__(256, 2)
void fp4_dequant_wmma_gemm(const float* __restrict__ X,     // [M_TOT, K_TOT] fp32
                           const int*   __restrict__ Widx,  // [N_TOT, K_TOT] int32 codes 0..15
                           const float* __restrict__ Amax,  // [N_TOT*K_TOT/QBLK]
                           const float* __restrict__ Code,  // [16]
                           const float* __restrict__ Bias,  // [N_TOT]
                           float*       __restrict__ Out)   // [M_TOT, N_TOT]
{
    // Ping-pong LDS tiles: one barrier per K-step instead of two.
    __shared__ __align__(16) _Float16 Xs[2][BM * LDSS];
    __shared__ __align__(16) _Float16 Ws[2][BN * LDSS];
    __shared__ float codeS[16];

    const int tid   = threadIdx.x;       // 0..255  (8 wave32 waves)
    const int lane  = tid & 31;
    const int wave  = tid >> 5;          // 0..7
    const int waveM = wave & 1;          // 2 waves along M -> 2*64 = 128
    const int waveN = wave >> 1;         // 4 waves along N -> 4*32 = 128
    const int bm0   = blockIdx.y * BM;
    const int bn0   = blockIdx.x * BN;

    if (tid < 16) codeS[tid] = Code[tid];
    __syncthreads();                     // codeS visible before first dequant

    v8f acc[4][2];
    const v8f zacc = {};
#pragma unroll
    for (int i = 0; i < 4; ++i)
#pragma unroll
        for (int j = 0; j < 2; ++j) acc[i][j] = zacc;

    // Per-thread tile-fill coordinates: each thread fills 32 contiguous elements
    // of one row (row = tid/2, half-row = tid%2).
    const int fr = tid >> 1;             // 0..127
    const int fc = (tid & 1) * 32;       // 0 or 32

    const float* xbase  = X    + (size_t)(bm0 + fr) * K_TOT + fc;
    const int*   wbase  = Widx + (size_t)(bn0 + fr) * K_TOT + fc;
    const float* ambase = Amax + (size_t)(bn0 + fr) * (K_TOT / QBLK);

    // Fragment-load coordinates (§7.12.2 layouts).
    const int aRowSel = lane & 15;
    const int aKsel   = (lane >> 4) * 8;   // lo lanes: K+0/K+16 ; hi lanes: K+8/K+24
    const int bColSel = lane & 15;
    const int bKsel   = (lane >> 4) * 8;   // lo lanes: K+0/K+16 ; hi lanes: K+8/K+24

    // ---- Software pipeline: stage tile k0=0 into registers ----------------
    f4v  xreg[8];
    i4v  wreg[8];
    float amreg = ambase[0];
#pragma unroll
    for (int j = 0; j < 8; ++j) {
        xreg[j] = *(const f4v*)(xbase + j * 4);          // global_load_b128
        wreg[j] = *(const i4v*)(wbase + j * 4);          // global_load_b128
    }

    for (int k0 = 0; k0 < K_TOT; k0 += BK) {
        const int buf = (k0 >> 6) & 1;

        // ---- Drain staged registers into LDS (fp32->f16, FP4 dequant) ----
        {
            _Float16* xdst = &Xs[buf][fr * LDSS + fc];
            _Float16* wdst = &Ws[buf][fr * LDSS + fc];
#pragma unroll
            for (int j = 0; j < 4; ++j) {               // 8 halves per store
                const f4v xa = xreg[2 * j], xb = xreg[2 * j + 1];
                v8h hx;
                hx[0] = (_Float16)xa.x;  hx[1] = (_Float16)xa.y;
                hx[2] = (_Float16)xa.z;  hx[3] = (_Float16)xa.w;
                hx[4] = (_Float16)xb.x;  hx[5] = (_Float16)xb.y;
                hx[6] = (_Float16)xb.z;  hx[7] = (_Float16)xb.w;
                *(v8h*)(xdst + j * 8) = hx;              // ds_store_b128

                const i4v wa = wreg[2 * j], wb = wreg[2 * j + 1];
                v8h hw;
                hw[0] = (_Float16)(codeS[wa.x] * amreg);
                hw[1] = (_Float16)(codeS[wa.y] * amreg);
                hw[2] = (_Float16)(codeS[wa.z] * amreg);
                hw[3] = (_Float16)(codeS[wa.w] * amreg);
                hw[4] = (_Float16)(codeS[wb.x] * amreg);
                hw[5] = (_Float16)(codeS[wb.y] * amreg);
                hw[6] = (_Float16)(codeS[wb.z] * amreg);
                hw[7] = (_Float16)(codeS[wb.w] * amreg);
                *(v8h*)(wdst + j * 8) = hw;              // ds_store_b128
            }
        }
        __syncthreads();   // tiles[buf] complete; (sole barrier per K-step)

        // ---- Kick off next tile's global loads; latency hides under WMMAs
        if (k0 + BK < K_TOT) {
            const float* xn = xbase + (k0 + BK);
            const int*   wn = wbase + (k0 + BK);
            amreg = ambase[(k0 + BK) >> 6];
#pragma unroll
            for (int j = 0; j < 8; ++j) {
                xreg[j] = *(const f4v*)(xn + j * 4);     // global_load_b128
                wreg[j] = *(const i4v*)(wn + j * 4);     // global_load_b128
            }
        }

        // ---- Compute: 2 K-substeps of 32, 8 WMMAs each -------------------
#pragma unroll
        for (int ks = 0; ks < BK; ks += 32) {
            v16h afrag[4];
#pragma unroll
            for (int i = 0; i < 4; ++i) {
                const int row = waveM * 64 + i * 16 + aRowSel;
                const _Float16* p = &Xs[buf][row * LDSS + ks + aKsel];
                v8h lo = *(const v8h*)(p);        // ds_load_b128  (K+0..7)
                v8h hi = *(const v8h*)(p + 16);   // ds_load_b128  (K+16..23)
                afrag[i] = __builtin_shufflevector(lo, hi,
                    0, 1, 2, 3, 4, 5, 6, 7, 8, 9, 10, 11, 12, 13, 14, 15);
            }
            v16h bfrag[2];
#pragma unroll
            for (int j = 0; j < 2; ++j) {
                const int col = waveN * 32 + j * 16 + bColSel;
                const _Float16* p = &Ws[buf][col * LDSS + ks + bKsel];
                v8h lo = *(const v8h*)(p);        // ds_load_b128  (K+0..7)
                v8h hi = *(const v8h*)(p + 16);   // ds_load_b128  (K+16..23)
                bfrag[j] = __builtin_shufflevector(lo, hi,
                    0, 1, 2, 3, 4, 5, 6, 7, 8, 9, 10, 11, 12, 13, 14, 15);
            }
#pragma unroll
            for (int i = 0; i < 4; ++i)
#pragma unroll
                for (int j = 0; j < 2; ++j)
                    acc[i][j] = __builtin_amdgcn_wmma_f32_16x16x32_f16(
                        /*neg_a=*/false, afrag[i],
                        /*neg_b=*/false, bfrag[j],
                        /*c_mod=*/(short)0, acc[i][j],
                        /*reuse_a=*/false, /*reuse_b=*/false);
        }
        // No trailing barrier: next iteration stores into the other buffer.
    }

    // ---- Epilogue: bias add + store (C/D layout: VGPR v -> rows v, v+8) ----
#pragma unroll
    for (int i = 0; i < 4; ++i) {
#pragma unroll
        for (int j = 0; j < 2; ++j) {
            const int row0 = bm0 + waveM * 64 + i * 16 + (lane >> 4) * 8;
            const int col  = bn0 + waveN * 32 + j * 16 + (lane & 15);
            const float b  = Bias[col];
#pragma unroll
            for (int v = 0; v < 8; ++v) {
                Out[(size_t)(row0 + v) * N_TOT + col] = acc[i][j][v] + b;
            }
        }
    }
}

extern "C" void kernel_launch(void* const* d_in, const int* in_sizes, int n_in,
                              void* d_out, int out_size, void* d_ws, size_t ws_size,
                              hipStream_t stream) {
    (void)in_sizes; (void)n_in; (void)out_size; (void)d_ws; (void)ws_size;
    const float* x    = (const float*)d_in[0];  // [4,2048,4096] fp32
    const int*   widx = (const int*)  d_in[1];  // [4096,4096] int32
    const float* amax = (const float*)d_in[2];  // [262144] fp32
    const float* code = (const float*)d_in[3];  // [16] fp32
    const float* bias = (const float*)d_in[4];  // [4096] fp32
    float*       out  = (float*)d_out;          // [4,2048,4096] fp32

    dim3 grid(N_TOT / BN, M_TOT / BM);          // (32, 64)
    dim3 block(256);
    fp4_dequant_wmma_gemm<<<grid, block, 0, stream>>>(x, widx, amax, code, bias, out);
}
